// TransformerLayer_41850161332806
// MI455X (gfx1250) — compile-verified, matrix operand order checked by reading, not run on hardware
//
#include <hip/hip_runtime.h>
#include <hip/hip_bf16.h>
#include <math.h>
#include <stdint.h>

// ---------------------------------------------------------------------------
// Problem constants (from reference)
// ---------------------------------------------------------------------------
#define BB 4
#define TT 1024
#define DD 1024
#define HH 16
#define DHH 64
#define DCC 64
#define NBUCK 32

typedef unsigned short ushort_t;
typedef __attribute__((ext_vector_type(16))) __bf16 v16bf;
typedef __attribute__((ext_vector_type(8)))  float  v8f;

union Frag {
    v16bf v;
    uint4 u4[2];
};

__device__ __forceinline__ v8f v8f_zero() {
    v8f z;
#pragma unroll
    for (int i = 0; i < 8; ++i) z[i] = 0.0f;
    return z;
}

__device__ __forceinline__ __bf16 f32_to_bf16(float f) {
    unsigned u = __builtin_bit_cast(unsigned, f);
    unsigned r = u + 0x7FFFu + ((u >> 16) & 1u);   // round-to-nearest-even
    unsigned short h = (unsigned short)(r >> 16);
    return __builtin_bit_cast(__bf16, h);
}

__device__ __forceinline__ float bf16_to_f32(__bf16 b) {
    unsigned short h = __builtin_bit_cast(unsigned short, b);
    unsigned u = ((unsigned)h) << 16;
    return __builtin_bit_cast(float, u);
}

__device__ __forceinline__ v8f wmma_bf16(v16bf a, v16bf b, v8f c) {
    // D = A(16x32 bf16) * B(32x16 bf16) + C(16x16 f32)
    return __builtin_amdgcn_wmma_f32_16x16x32_bf16(
        /*neg_a=*/false, a, /*neg_b=*/false, b,
        /*c_mod=*/(short)0, c, /*reuse_a=*/false, /*reuse_b=*/false);
}

// Async copy 16B (8 bf16) from global to LDS; tracked with ASYNCcnt.
__device__ __forceinline__ void async_copy_b128(const void* gsrc, void* lds_dst) {
    unsigned loff = (unsigned)(unsigned long long)lds_dst;  // LDS aperture: low 32 bits
    asm volatile("global_load_async_to_lds_b128 %0, %1, off"
                 :: "v"(loff), "v"(gsrc) : "memory");
}
__device__ __forceinline__ void async_wait0() {
    asm volatile("s_wait_asynccnt 0x0" ::: "memory");
}

// ---------------------------------------------------------------------------
// Elementwise helpers
// ---------------------------------------------------------------------------
__global__ void cvt_bf16_kernel(const float* __restrict__ src,
                                __bf16* __restrict__ dst, int n) {
    int i = blockIdx.x * blockDim.x + threadIdx.x;
    if (i < n) dst[i] = f32_to_bf16(src[i]);
}

// gamma/beta = cond @ W + b  (per batch, DC=64 reduction)
__global__ void film_params_kernel(const float* __restrict__ cond,
                                   const float* __restrict__ gw,
                                   const float* __restrict__ gb,
                                   const float* __restrict__ bw,
                                   const float* __restrict__ bb,
                                   float* __restrict__ gamma,
                                   float* __restrict__ beta) {
    int idx = blockIdx.x * blockDim.x + threadIdx.x;   // B*D
    if (idx >= BB * DD) return;
    int b = idx / DD, d = idx % DD;
    const float* c = cond + b * DCC;
    float g = gb[d], bt = bb[d];
#pragma unroll 8
    for (int i = 0; i < DCC; ++i) {
        float cv = c[i];
        g  += cv * gw[i * DD + d];
        bt += cv * bw[i * DD + d];
    }
    gamma[idx] = g;
    beta[idx]  = bt;
}

// T5 relative position bias -> bias_out[h][q][k]  (this is output #2)
__global__ void bias_kernel(const float* __restrict__ rel_emb,
                            float* __restrict__ bias_out, int n) {
    int idx = blockIdx.x * blockDim.x + threadIdx.x;
    if (idx >= n) return;
    int k = idx % TT;
    int q = (idx / TT) % TT;
    int h = idx / (TT * TT);
    int rp = k - q;
    const int nb = NBUCK / 2;      // 16
    const int max_exact = nb / 2;  // 8
    int bucket = (rp > 0) ? nb : 0;
    int arp = rp < 0 ? -rp : rp;
    int val;
    if (arp < max_exact) {
        val = arp;
    } else {
        float rp_safe = (float)(arp > 1 ? arp : 1);
        int large = max_exact +
            (int)(logf(rp_safe / (float)max_exact) /
                  logf(128.0f / (float)max_exact) * (float)(nb - max_exact));
        val = large < nb - 1 ? large : nb - 1;
    }
    bucket += val;
    bias_out[idx] = rel_emb[bucket * HH + h];
}

// y = film(rms_norm(x)) -> bf16, one block per (b,t) row
__global__ __launch_bounds__(256)
void rmsnorm_film_kernel(const float* __restrict__ x,
                         const float* __restrict__ w,
                         const float* __restrict__ gamma,
                         const float* __restrict__ beta,
                         __bf16* __restrict__ y) {
    int row = blockIdx.x;          // 0 .. B*T-1
    int b = row / TT;
    const float* xr = x + (size_t)row * DD;
    float ss = 0.0f;
    for (int i = threadIdx.x; i < DD; i += 256) {
        float v = xr[i];
        ss += v * v;
    }
#pragma unroll
    for (int m = 16; m >= 1; m >>= 1) ss += __shfl_xor(ss, m, 32);
    __shared__ float red[8];
    if ((threadIdx.x & 31) == 0) red[threadIdx.x >> 5] = ss;
    __syncthreads();
    float tot = 0.0f;
#pragma unroll
    for (int i = 0; i < 8; ++i) tot += red[i];
    float rstd = rsqrtf(tot / (float)DD + 1e-6f);
    const float* g = gamma + (size_t)b * DD;
    const float* be = beta + (size_t)b * DD;
    __bf16* yr = y + (size_t)row * DD;
    for (int i = threadIdx.x; i < DD; i += 256) {
        float v = w[i] * (xr[i] * rstd);
        v = v * (g[i] + 1.0f) + be[i];
        yr[i] = f32_to_bf16(v);
    }
}

// geglu: g[r,c] = h[r,c] * gelu(h[r,c+2048]), exact gelu
__global__ void geglu_kernel(const __bf16* __restrict__ h,
                             __bf16* __restrict__ g, int n) {
    int idx = blockIdx.x * blockDim.x + threadIdx.x;  // rows * 2048
    if (idx >= n) return;
    int r = idx / (2 * DD);
    int c = idx % (2 * DD);
    float p1 = bf16_to_f32(h[(size_t)r * 4 * DD + c]);
    float p2 = bf16_to_f32(h[(size_t)r * 4 * DD + 2 * DD + c]);
    float ge = 0.5f * p2 * (1.0f + erff(p2 * 0.70710678118654752f));
    g[idx] = f32_to_bf16(p1 * ge);
}

// ---------------------------------------------------------------------------
// Generic bf16 WMMA GEMM:  C[M,N] = A[M,K] * B[K,N]  (row-major, bf16 in)
// Block tile 128x128, 8 waves (4 along M x 2 along N), wave tile 32x64.
// Double-buffered LDS; A tile staged via GLOBAL_LOAD_ASYNC_TO_LDS_B128,
// B tile transposed through VGPRs (needed for [n][k] fragment layout).
// EPI==0: store bf16.  EPI==1: store f32 with residual add.
// ---------------------------------------------------------------------------
template <int EPI>
__global__ __launch_bounds__(256)
void gemm_bf16_kernel(const __bf16* __restrict__ A,
                      const __bf16* __restrict__ Bm,
                      void* __restrict__ C,
                      const float* __restrict__ residual,
                      int M, int N, int K) {
    __shared__ ushort_t lA[2][128 * 32];   // [m][k]
    __shared__ ushort_t lB[2][128 * 32];   // [n][k]  (B transposed into LDS)

    const int tid  = threadIdx.x;
    const int lane = tid & 31;
    const int wave = tid >> 5;          // 0..7
    const int wm   = wave & 3;          // 4 waves along M
    const int wn   = wave >> 2;         // 2 waves along N
    const int half = lane >> 4;         // which 16-lane half
    const int l16  = lane & 15;
    const int koff = half ? 8 : 0;

    const int blockM = blockIdx.y * 128;
    const int blockN = blockIdx.x * 128;

    // per-thread tile-load coordinates (512 x 16B chunks per tile, 2 each)
    const int aRow0 = (tid + 0)   >> 2, aKc0 = ((tid + 0)   & 3) * 8;
    const int aRow1 = (tid + 256) >> 2, aKc1 = ((tid + 256) & 3) * 8;
    const int bKr0  = (tid + 0)   >> 4, bNc0 = ((tid + 0)   & 15) * 8;
    const int bKr1  = (tid + 256) >> 4, bNc1 = ((tid + 256) & 15) * 8;

    auto issueA = [&](int buf, int k0) {
        async_copy_b128(A + (size_t)(blockM + aRow0) * K + k0 + aKc0,
                        &lA[buf][aRow0 * 32 + aKc0]);
        async_copy_b128(A + (size_t)(blockM + aRow1) * K + k0 + aKc1,
                        &lA[buf][aRow1 * 32 + aKc1]);
    };
    auto loadB = [&](int k0, uint4* st) {
        st[0] = *(const uint4*)(Bm + (size_t)(k0 + bKr0) * N + blockN + bNc0);
        st[1] = *(const uint4*)(Bm + (size_t)(k0 + bKr1) * N + blockN + bNc1);
    };
    auto storeB = [&](int buf, const uint4* st) {
        const ushort_t* d0 = (const ushort_t*)&st[0];
        const ushort_t* d1 = (const ushort_t*)&st[1];
#pragma unroll
        for (int e = 0; e < 8; ++e) lB[buf][(bNc0 + e) * 32 + bKr0] = d0[e];
#pragma unroll
        for (int e = 0; e < 8; ++e) lB[buf][(bNc1 + e) * 32 + bKr1] = d1[e];
    };

    v8f acc[2][4];
#pragma unroll
    for (int i = 0; i < 2; ++i)
#pragma unroll
        for (int j = 0; j < 4; ++j) acc[i][j] = v8f_zero();

    // prologue: stage tile 0
    {
        uint4 bst[2];
        issueA(0, 0);
        loadB(0, bst);
        storeB(0, bst);
        async_wait0();
        __syncthreads();
    }

    int cur = 0;
    for (int k0 = 0; k0 < K; k0 += 32) {
        const int nxt = cur ^ 1;
        const bool has_next = (k0 + 32) < K;
        uint4 bst[2];
        if (has_next) {
            issueA(nxt, k0 + 32);       // async: overlaps with compute below
            loadB(k0 + 32, bst);        // global->VGPR, consumed after compute
        }

        // compute current tile
        Frag a[2], b[4];
#pragma unroll
        for (int i = 0; i < 2; ++i) {
            int row = wm * 32 + i * 16 + l16;
            a[i].u4[0] = *(const uint4*)&lA[cur][row * 32 + koff];
            a[i].u4[1] = *(const uint4*)&lA[cur][row * 32 + koff + 16];
        }
#pragma unroll
        for (int j = 0; j < 4; ++j) {
            int col = wn * 64 + j * 16 + l16;
            b[j].u4[0] = *(const uint4*)&lB[cur][col * 32 + koff];
            b[j].u4[1] = *(const uint4*)&lB[cur][col * 32 + koff + 16];
        }
#pragma unroll
        for (int i = 0; i < 2; ++i)
#pragma unroll
            for (int j = 0; j < 4; ++j)
                acc[i][j] = wmma_bf16(a[i].v, b[j].v, acc[i][j]);

        if (has_next) storeB(nxt, bst);
        async_wait0();
        __syncthreads();
        cur = nxt;
    }

    // epilogue (C-layout: col = l16, row = r + 8*half per vgpr r)
#pragma unroll
    for (int i = 0; i < 2; ++i) {
#pragma unroll
        for (int j = 0; j < 4; ++j) {
            int col = blockN + wn * 64 + j * 16 + l16;
#pragma unroll
            for (int r = 0; r < 8; ++r) {
                int row = blockM + wm * 32 + i * 16 + r + half * 8;
                size_t idx = (size_t)row * N + col;
                float v = acc[i][j][r];
                if (EPI == 0) {
                    ((__bf16*)C)[idx] = f32_to_bf16(v);
                } else {
                    ((float*)C)[idx] = residual[idx] + v;
                }
            }
        }
    }
}

// ---------------------------------------------------------------------------
// Flash attention: grid (B*H, T/64), block 128 (4 waves x 16 query rows).
// S = Q*K^T * 1/8 + bias, mask, online softmax, O = P*V.
// K tile staged async to LDS; V tile transposed through VGPRs.
// ---------------------------------------------------------------------------
__global__ __launch_bounds__(128)
void flash_attn_kernel(const __bf16* __restrict__ qb,
                       const __bf16* __restrict__ kb,
                       const __bf16* __restrict__ vb,
                       const float* __restrict__ bias,   // [H][T][T]
                       const int* __restrict__ mask,     // [B][T][T]
                       __bf16* __restrict__ ob) {
    __shared__ ushort_t lK[64 * 64];   // [key][dh]   -> B-frag layout for Q*K^T
    __shared__ ushort_t lV[64 * 64];   // [dh][key]   -> B-frag layout for P*V
    __shared__ ushort_t lP[64 * 64];   // [q][key]    -> A-frag relayout

    const int bh = blockIdx.x;
    const int b = bh / HH;
    const int h = bh % HH;
    const int qbase = blockIdx.y * 64;
    const int tid  = threadIdx.x;
    const int lane = tid & 31;
    const int wave = tid >> 5;         // 0..3 -> 16 query rows each
    const int half = lane >> 4;
    const int l16  = lane & 15;
    const int koff = half ? 8 : 0;

    // Q fragments stay in registers for the whole kernel
    Frag qf[2];
    {
        int qrow = qbase + wave * 16 + l16;
        const __bf16* qrp = qb + ((size_t)(b * TT + qrow)) * DD + h * DHH;
        qf[0].u4[0] = *(const uint4*)(qrp + koff);
        qf[0].u4[1] = *(const uint4*)(qrp + koff + 16);
        qf[1].u4[0] = *(const uint4*)(qrp + 32 + koff);
        qf[1].u4[1] = *(const uint4*)(qrp + 32 + koff + 16);
    }

    float m[8], l[8];
    v8f o[4];
#pragma unroll
    for (int r = 0; r < 8; ++r) { m[r] = -1e30f; l[r] = 0.0f; }
#pragma unroll
    for (int j = 0; j < 4; ++j) o[j] = v8f_zero();

    const float scale = 0.125f;   // 1/sqrt(DH)

    for (int kt = 0; kt < TT / 64; ++kt) {
        const int kbase = kt * 64;
        // K tile: natural [key][dh] layout, async to LDS (no VGPR staging)
#pragma unroll
        for (int i = 0; i < 4; ++i) {
            int chunk = tid + i * 128;
            int row = chunk >> 3;
            int cc  = (chunk & 7) * 8;
            async_copy_b128(kb + ((size_t)(b * TT + kbase + row)) * DD + h * DHH + cc,
                            &lK[row * 64 + cc]);
        }
        // V tile transposed: lV[dh][key]
#pragma unroll
        for (int i = 0; i < 4; ++i) {
            int chunk = tid + i * 128;
            int row = chunk >> 3;
            int cc  = (chunk & 7) * 8;
            uint4 d = *(const uint4*)(vb +
                ((size_t)(b * TT + kbase + row)) * DD + h * DHH + cc);
            const ushort_t* ds = (const ushort_t*)&d;
#pragma unroll
            for (int e = 0; e < 8; ++e) lV[(cc + e) * 64 + row] = ds[e];
        }
        async_wait0();
        __syncthreads();

        // S = Q * K^T  (4 column frags of 16, K-reduction DH=64 -> 2 wmma)
        v8f s[4];
#pragma unroll
        for (int j = 0; j < 4; ++j) {
            int col = j * 16 + l16;
            Frag b0, b1;
            b0.u4[0] = *(const uint4*)&lK[col * 64 + koff];
            b0.u4[1] = *(const uint4*)&lK[col * 64 + koff + 16];
            b1.u4[0] = *(const uint4*)&lK[col * 64 + 32 + koff];
            b1.u4[1] = *(const uint4*)&lK[col * 64 + 32 + koff + 16];
            v8f z = v8f_zero();
            z = wmma_bf16(qf[0].v, b0.v, z);
            s[j] = wmma_bf16(qf[1].v, b1.v, z);
        }

        // scale + bias + mask + online softmax (per accumulator row)
#pragma unroll
        for (int r = 0; r < 8; ++r) {
            int qg = qbase + wave * 16 + r + half * 8;
            float rmax = -1e30f;
#pragma unroll
            for (int j = 0; j < 4; ++j) {
                int kg = kbase + j * 16 + l16;
                float v = s[j][r] * scale +
                          bias[((size_t)h * TT + qg) * TT + kg];
                int mk = mask[((size_t)b * TT + qg) * TT + kg];
                v = (mk == 0) ? -1e9f : v;
                s[j][r] = v;
                rmax = fmaxf(rmax, v);
            }
#pragma unroll
            for (int mm = 8; mm >= 1; mm >>= 1)
                rmax = fmaxf(rmax, __shfl_xor(rmax, mm, 32));
            float mnew = fmaxf(m[r], rmax);
            float corr = __expf(m[r] - mnew);
            float psum = 0.0f;
#pragma unroll
            for (int j = 0; j < 4; ++j) {
                float p = __expf(s[j][r] - mnew);
                s[j][r] = p;
                psum += p;
            }
#pragma unroll
            for (int mm = 8; mm >= 1; mm >>= 1)
                psum += __shfl_xor(psum, mm, 32);
            l[r] = l[r] * corr + psum;
            m[r] = mnew;
#pragma unroll
            for (int j = 0; j < 4; ++j) o[j][r] *= corr;
        }

        // Relayout P through LDS (C-frag -> A-frag)
#pragma unroll
        for (int j = 0; j < 4; ++j) {
            int coll = j * 16 + l16;
#pragma unroll
            for (int r = 0; r < 8; ++r) {
                int rowl = wave * 16 + r + half * 8;
                lP[rowl * 64 + coll] =
                    __builtin_bit_cast(unsigned short, f32_to_bf16(s[j][r]));
            }
        }
        __syncthreads();

        // O += P * V
        Frag pa[2];
        {
            int prow = wave * 16 + l16;
            pa[0].u4[0] = *(const uint4*)&lP[prow * 64 + koff];
            pa[0].u4[1] = *(const uint4*)&lP[prow * 64 + koff + 16];
            pa[1].u4[0] = *(const uint4*)&lP[prow * 64 + 32 + koff];
            pa[1].u4[1] = *(const uint4*)&lP[prow * 64 + 32 + koff + 16];
        }
#pragma unroll
        for (int j = 0; j < 4; ++j) {
            int col = j * 16 + l16;   // dh column
            Frag v0, v1;
            v0.u4[0] = *(const uint4*)&lV[col * 64 + koff];
            v0.u4[1] = *(const uint4*)&lV[col * 64 + koff + 16];
            v1.u4[0] = *(const uint4*)&lV[col * 64 + 32 + koff];
            v1.u4[1] = *(const uint4*)&lV[col * 64 + 32 + koff + 16];
            o[j] = wmma_bf16(pa[0].v, v0.v, o[j]);
            o[j] = wmma_bf16(pa[1].v, v1.v, o[j]);
        }
        __syncthreads();
    }

    // finalize: O /= l, write bf16 [B,T,H*DH]
#pragma unroll
    for (int j = 0; j < 4; ++j) {
        int dh = j * 16 + l16;
#pragma unroll
        for (int r = 0; r < 8; ++r) {
            int qg = qbase + wave * 16 + r + half * 8;
            float v = o[j][r] / l[r];
            ob[((size_t)(b * TT + qg)) * DD + h * DHH + dh] = f32_to_bf16(v);
        }
    }
}

// ---------------------------------------------------------------------------
// Host-side orchestration
// ---------------------------------------------------------------------------
static inline size_t align256(size_t x) { return (x + 255) & ~(size_t)255; }

extern "C" void kernel_launch(void* const* d_in, const int* in_sizes, int n_in,
                              void* d_out, int out_size, void* d_ws, size_t ws_size,
                              hipStream_t stream) {
    (void)in_sizes; (void)n_in; (void)out_size; (void)ws_size;
    const float* x        = (const float*)d_in[0];
    const int*   x_mask   = (const int*)  d_in[1];
    const float* cond     = (const float*)d_in[2];
    const float* norm1_w  = (const float*)d_in[3];
    const float* f1_gw    = (const float*)d_in[4];
    const float* f1_gb    = (const float*)d_in[5];
    const float* f1_bw    = (const float*)d_in[6];
    const float* f1_bb    = (const float*)d_in[7];
    const float* wq       = (const float*)d_in[8];
    const float* wk       = (const float*)d_in[9];
    const float* wv       = (const float*)d_in[10];
    const float* wo       = (const float*)d_in[11];
    const float* rel_emb  = (const float*)d_in[12];
    const float* norm3_w  = (const float*)d_in[13];
    const float* f3_gw    = (const float*)d_in[14];
    const float* f3_gb    = (const float*)d_in[15];
    const float* f3_bw    = (const float*)d_in[16];
    const float* f3_bb    = (const float*)d_in[17];
    const float* ffn_w1   = (const float*)d_in[18];
    const float* ffn_w2   = (const float*)d_in[19];

    float* out_x    = (float*)d_out;                              // [B,T,D]
    float* out_bias = (float*)d_out + (size_t)BB * TT * DD;       // [H,T,T]

    // workspace bump allocator
    char* ws = (char*)d_ws;
    size_t off = 0;
    auto alloc = [&](size_t bytes) -> void* {
        void* p = ws + off;
        off += align256(bytes);
        return p;
    };
    const size_t BT = (size_t)BB * TT;
    __bf16* wq_bf   = (__bf16*)alloc((size_t)DD * DD * 2);
    __bf16* wk_bf   = (__bf16*)alloc((size_t)DD * DD * 2);
    __bf16* wv_bf   = (__bf16*)alloc((size_t)DD * DD * 2);
    __bf16* wo_bf   = (__bf16*)alloc((size_t)DD * DD * 2);
    __bf16* w1_bf   = (__bf16*)alloc((size_t)DD * 4 * DD * 2);
    __bf16* w2_bf   = (__bf16*)alloc((size_t)2 * DD * DD * 2);
    float*  gamma1  = (float*)alloc((size_t)BB * DD * 4);
    float*  beta1   = (float*)alloc((size_t)BB * DD * 4);
    float*  gamma3  = (float*)alloc((size_t)BB * DD * 4);
    float*  beta3   = (float*)alloc((size_t)BB * DD * 4);
    __bf16* y_bf    = (__bf16*)alloc(BT * DD * 2);
    __bf16* q_bf    = (__bf16*)alloc(BT * DD * 2);
    __bf16* k_bf    = (__bf16*)alloc(BT * DD * 2);
    __bf16* v_bf    = (__bf16*)alloc(BT * DD * 2);
    __bf16* attn_bf = (__bf16*)alloc(BT * DD * 2);
    float*  x_mid   = (float*)alloc(BT * DD * 4);
    __bf16* y3_bf   = (__bf16*)alloc(BT * DD * 2);
    __bf16* h_bf    = (__bf16*)alloc(BT * 4 * DD * 2);
    __bf16* g_bf    = (__bf16*)alloc(BT * 2 * DD * 2);

    // 1) weight conversion fp32 -> bf16
    {
        int n = DD * DD;
        cvt_bf16_kernel<<<(n + 255) / 256, 256, 0, stream>>>(wq, wq_bf, n);
        cvt_bf16_kernel<<<(n + 255) / 256, 256, 0, stream>>>(wk, wk_bf, n);
        cvt_bf16_kernel<<<(n + 255) / 256, 256, 0, stream>>>(wv, wv_bf, n);
        cvt_bf16_kernel<<<(n + 255) / 256, 256, 0, stream>>>(wo, wo_bf, n);
        int n1 = DD * 4 * DD;
        cvt_bf16_kernel<<<(n1 + 255) / 256, 256, 0, stream>>>(ffn_w1, w1_bf, n1);
        int n2 = 2 * DD * DD;
        cvt_bf16_kernel<<<(n2 + 255) / 256, 256, 0, stream>>>(ffn_w2, w2_bf, n2);
    }
    // 2) FiLM params
    film_params_kernel<<<(BB * DD + 255) / 256, 256, 0, stream>>>(
        cond, f1_gw, f1_gb, f1_bw, f1_bb, gamma1, beta1);
    film_params_kernel<<<(BB * DD + 255) / 256, 256, 0, stream>>>(
        cond, f3_gw, f3_gb, f3_bw, f3_bb, gamma3, beta3);
    // 3) position bias (output #2 + consumed by attention)
    {
        int n = HH * TT * TT;
        bias_kernel<<<(n + 255) / 256, 256, 0, stream>>>(rel_emb, out_bias, n);
    }
    // 4) y = film(rmsnorm(x)) -> bf16
    rmsnorm_film_kernel<<<(int)BT, 256, 0, stream>>>(x, norm1_w, gamma1, beta1, y_bf);
    // 5) Q,K,V projections
    {
        dim3 grid(DD / 128, (int)(BT / 128));
        gemm_bf16_kernel<0><<<grid, 256, 0, stream>>>(y_bf, wq_bf, q_bf, nullptr,
                                                      (int)BT, DD, DD);
        gemm_bf16_kernel<0><<<grid, 256, 0, stream>>>(y_bf, wk_bf, k_bf, nullptr,
                                                      (int)BT, DD, DD);
        gemm_bf16_kernel<0><<<grid, 256, 0, stream>>>(y_bf, wv_bf, v_bf, nullptr,
                                                      (int)BT, DD, DD);
    }
    // 6) flash attention
    {
        dim3 grid(BB * HH, TT / 64);
        flash_attn_kernel<<<grid, 128, 0, stream>>>(q_bf, k_bf, v_bf,
                                                    out_bias, x_mask, attn_bf);
    }
    // 7) x_mid = x + attn @ wo
    {
        dim3 grid(DD / 128, (int)(BT / 128));
        gemm_bf16_kernel<1><<<grid, 256, 0, stream>>>(attn_bf, wo_bf, x_mid, x,
                                                      (int)BT, DD, DD);
    }
    // 8) y3 = film(rmsnorm(x_mid)) -> bf16
    rmsnorm_film_kernel<<<(int)BT, 256, 0, stream>>>(x_mid, norm3_w, gamma3, beta3, y3_bf);
    // 9) h = y3 @ ffn_w1  [BT, 4D]
    {
        dim3 grid(4 * DD / 128, (int)(BT / 128));
        gemm_bf16_kernel<0><<<grid, 256, 0, stream>>>(y3_bf, w1_bf, h_bf, nullptr,
                                                      (int)BT, 4 * DD, DD);
    }
    // 10) geglu
    {
        int n = (int)(BT * 2 * DD);
        geglu_kernel<<<(n + 255) / 256, 256, 0, stream>>>(h_bf, g_bf, n);
    }
    // 11) out = x_mid + g @ ffn_w2
    {
        dim3 grid(DD / 128, (int)(BT / 128));
        gemm_bf16_kernel<1><<<grid, 256, 0, stream>>>(g_bf, w2_bf, out_x, x_mid,
                                                      (int)BT, DD, 2 * DD);
    }
}
